// TransformerEncoder_16724602651385
// MI455X (gfx1250) — matrix operand order repaired and checked by reading, not compiled, not run on hardware
//
#include <hip/hip_runtime.h>
#include <hip/hip_bf16.h>
#include <stdint.h>

typedef __bf16 bf16;
typedef __bf16 v16bf __attribute__((ext_vector_type(16)));
typedef float  v8f   __attribute__((ext_vector_type(8)));
typedef int    v4i   __attribute__((ext_vector_type(4)));

#define D_ 1024
#define S_ 4096
#define B_ 4
#define N_ (B_*S_)   // 16384 rows total

// gfx1250 async global->LDS staging (ASYNCcnt), if the toolchain exposes it.
#if __has_builtin(__builtin_amdgcn_global_load_async_to_lds_b128) && \
    __has_builtin(__builtin_amdgcn_s_wait_asynccnt)
#define USE_ASYNC_LDS 1
typedef __attribute__((address_space(1))) v4i* gv4i_t;   // global int4*
typedef __attribute__((address_space(3))) v4i* lv4i_t;   // LDS int4*
#else
#define USE_ASYNC_LDS 0
#endif

// Build a 16-element bf16 fragment from two 16-byte chunks.
// A-fragments (16x32): chunks at +0 and +16 elements (per CDNA5 A layout:
//   lane<16 -> M=lane, K = {k0..k0+7} U {16+k0..16+k0+7}, k0 = (lane>>4)*8).
// B-fragments (32x16): same K order as A -> chunks at +0 and +16.
static __device__ __forceinline__ v16bf load_frag(const bf16* p0, const bf16* p1) {
    union { v16bf v; uint4 q[2]; } u;
    u.q[0] = *reinterpret_cast<const uint4*>(p0);
    u.q[1] = *reinterpret_cast<const uint4*>(p1);
    return u.v;
}

__global__ void cvt_f32_bf16(const float* __restrict__ src, bf16* __restrict__ dst, int n) {
    int i = blockIdx.x * blockDim.x + threadIdx.x;
    if (i < n) dst[i] = (bf16)src[i];
}

// ---------------------------------------------------------------------------
// Tiled GEMM: C[N x 1024] = A[N x 1024] * W[1024 x 1024], epilogue by MODE.
//   MODE 0: out bf16 = acc + bias                       (Q, K projections)
//   MODE 1: out bf16 = relu(acc + bias)                 (FFN1)
//   MODE 2: out f32  = 2*(acc + bias)                   (FFN2 -> d_out)
//   MODE 3: out bf16 transposed to [b][D][S] layout     (V^T for attention)
// Block: 256 threads = 8 waves in a 2x4 grid; each wave owns 64x32 of the
// 128x128 C tile (4x2 accumulators of v_wmma_f32_16x16x32_bf16).
// A tile is staged memory->LDS via global_load_async_to_lds_b128 when the
// builtin exists (no VGPR round-trip); B tile is K-transposed through VGPRs.
// ---------------------------------------------------------------------------
template<int MODE>
__global__ __launch_bounds__(256)
void gemm128(const bf16* __restrict__ A, const bf16* __restrict__ W,
             const float* __restrict__ bias, void* __restrict__ out)
{
    __shared__ bf16 As[2][128 * 32];   // row-major 128x32 tile of A
    __shared__ bf16 Bs[2][128 * 32];   // K-transposed: Bs[n][k], 128 cols x 32 k
    const int tid  = threadIdx.x;
    const int lane = tid & 31;
    const int wave = tid >> 5;
    const int wm = wave >> 2, wn = wave & 3;
    const int rowbase = blockIdx.y * 128;
    const int colbase = blockIdx.x * 128;

    v8f zero = {};
    v8f acc[4][2];
#pragma unroll
    for (int mi = 0; mi < 4; ++mi)
#pragma unroll
        for (int ni = 0; ni < 2; ++ni) acc[mi][ni] = zero;

    // cooperative loader assignments
    const int la_row = tid >> 1;          // 0..127
    const int la_col = (tid & 1) * 16;    // 0 or 16
    const int lb_k   = tid >> 3;          // 0..31
    const int lb_n   = (tid & 7) * 16;    // 0,16,..,112

    auto load_tiles = [&](int kt, int buf) {
        const int k0 = kt * 32;
        const bf16* ap = A + (size_t)(rowbase + la_row) * D_ + k0 + la_col;
        bf16* asd = &As[buf][la_row * 32 + la_col];
#if USE_ASYNC_LDS
        __builtin_amdgcn_global_load_async_to_lds_b128(
            (gv4i_t)const_cast<bf16*>(ap),     (lv4i_t)asd,       0, 0);
        __builtin_amdgcn_global_load_async_to_lds_b128(
            (gv4i_t)const_cast<bf16*>(ap + 8), (lv4i_t)(asd + 8), 0, 0);
#else
        *reinterpret_cast<uint4*>(asd)     = *reinterpret_cast<const uint4*>(ap);
        *reinterpret_cast<uint4*>(asd + 8) = *reinterpret_cast<const uint4*>(ap + 8);
#endif
        const bf16* wp = W + (size_t)(k0 + lb_k) * D_ + colbase + lb_n;
        union { uint4 q; bf16 e[8]; } t0, t1;
        t0.q = *reinterpret_cast<const uint4*>(wp);
        t1.q = *reinterpret_cast<const uint4*>(wp + 8);
#pragma unroll
        for (int i = 0; i < 8; ++i) {
            Bs[buf][(lb_n + i)     * 32 + lb_k] = t0.e[i];
            Bs[buf][(lb_n + 8 + i) * 32 + lb_k] = t1.e[i];
        }
    };

    load_tiles(0, 0);
#if USE_ASYNC_LDS
    __builtin_amdgcn_s_wait_asynccnt(0);
#endif
    __syncthreads();

    const int half = lane >> 4;
    const int fm   = lane & 15;
    const int k0f  = half * 8;

    for (int kt = 0; kt < 32; ++kt) {
        const int cur = kt & 1;
        if (kt + 1 < 32) {
            __builtin_prefetch(A + (size_t)(rowbase + la_row) * D_ + (kt + 1) * 32, 0, 1);
            load_tiles(kt + 1, cur ^ 1);
        }
        v16bf afr[4], bfr[2];
#pragma unroll
        for (int mi = 0; mi < 4; ++mi) {
            const bf16* p = &As[cur][(wm * 64 + mi * 16 + fm) * 32 + k0f];
            afr[mi] = load_frag(p, p + 16);
        }
#pragma unroll
        for (int ni = 0; ni < 2; ++ni) {
            const bf16* p = &Bs[cur][(wn * 32 + ni * 16 + fm) * 32 + k0f];
            bfr[ni] = load_frag(p, p + 16);
        }
#pragma unroll
        for (int mi = 0; mi < 4; ++mi)
#pragma unroll
            for (int ni = 0; ni < 2; ++ni)
                acc[mi][ni] = __builtin_amdgcn_wmma_f32_16x16x32_bf16(
                    false, afr[mi], false, bfr[ni], (short)0, acc[mi][ni], false, false);
#if USE_ASYNC_LDS
        __builtin_amdgcn_s_wait_asynccnt(0);
#endif
        __syncthreads();
    }

    // Epilogue. C layout: element e of lane l -> row = e + (l>>4)*8, col = l&15.
#pragma unroll
    for (int mi = 0; mi < 4; ++mi) {
#pragma unroll
        for (int ni = 0; ni < 2; ++ni) {
#pragma unroll
            for (int e = 0; e < 8; ++e) {
                const int grow = rowbase + wm * 64 + mi * 16 + half * 8 + e;
                const int gcol = colbase + wn * 32 + ni * 16 + fm;
                float v = acc[mi][ni][e] + bias[gcol];
                if (MODE == 0) {
                    ((bf16*)out)[(size_t)grow * D_ + gcol] = (bf16)v;
                } else if (MODE == 1) {
                    ((bf16*)out)[(size_t)grow * D_ + gcol] = (bf16)fmaxf(v, 0.0f);
                } else if (MODE == 2) {
                    ((float*)out)[(size_t)grow * D_ + gcol] = 2.0f * v;
                } else { // MODE 3: transpose to [b][D][S]
                    const int bb = grow >> 12;          // /S_
                    const int ss = grow & (S_ - 1);
                    ((bf16*)out)[((size_t)bb * D_ + gcol) * S_ + ss] = (bf16)v;
                }
            }
        }
    }
}

// ---------------------------------------------------------------------------
// Flash attention: O = softmax(Q K^T / 32) V, plus residual x, output bf16.
// Block = 512 threads = 16 waves; block handles 16 query rows of one batch.
// Wave w owns output columns [w*64, w*64+64). Q fragments stay in registers
// for the whole pass. All K and V fragments for a 32-key tile are preloaded
// into distinct registers up front so the load clauses overlap the reduction
// and softmax barriers instead of stalling each WMMA on s_wait_loadcnt 0.
// ---------------------------------------------------------------------------
__global__ __launch_bounds__(512)
void flash_attn(const bf16* __restrict__ q, const bf16* __restrict__ k,
                const bf16* __restrict__ vt, const float* __restrict__ x,
                bf16* __restrict__ xa)
{
    __shared__ float red[16 * 256];     // per-wave 16x16 f32 score partials
    __shared__ float Stile[16][32];     // reduced scores for 32-key tile
    __shared__ bf16  Ps[16][32];        // probabilities (A-matrix for P*V)
    __shared__ float mrow[16], lrow[16], arow[16];

    const int tid  = threadIdx.x;
    const int lane = tid & 31;
    const int wave = tid >> 5;          // 0..15
    const int dw   = wave * 64;         // this wave's D-slice
    const int b    = blockIdx.x >> 8;
    const int q0   = (blockIdx.x & 255) * 16;
    const int half = lane >> 4;
    const int fm   = lane & 15;
    const int k0f  = half * 8;
    const int kb2  = half * 16;

    if (tid < 16) { mrow[tid] = -1e30f; lrow[tid] = 0.0f; }

    // resident Q fragments: rows q0..q0+15, K-dim = dw..dw+63 (2 frags)
    v16bf qa[2];
#pragma unroll
    for (int c = 0; c < 2; ++c) {
        const bf16* p = q + ((size_t)(b * S_ + q0 + fm)) * D_ + dw + c * 32 + k0f;
        qa[c] = load_frag(p, p + 16);
    }
    v8f zero = {};
    v8f acc[4];
#pragma unroll
    for (int t = 0; t < 4; ++t) acc[t] = zero;
    __syncthreads();

    for (int kt = 0; kt < S_ / 32; ++kt) {
        const int kb = kt * 32;

        // ---- preload every K and V fragment for this key tile ----
        v16bf kf[2][2];
#pragma unroll
        for (int ni = 0; ni < 2; ++ni)
#pragma unroll
            for (int c = 0; c < 2; ++c) {
                const bf16* p = k + ((size_t)(b * S_ + kb + ni * 16 + fm)) * D_
                              + dw + c * 32 + k0f;
                kf[ni][c] = load_frag(p, p + 16);
            }
        v16bf vf[4];
#pragma unroll
        for (int t = 0; t < 4; ++t) {
            const bf16* p = vt + ((size_t)(b * D_ + dw + t * 16 + fm)) * S_ + kb + k0f;
            vf[t] = load_frag(p, p + 16);
        }

        // ---- scores: two 16-key halves, each reduced across the 16 waves ----
#pragma unroll
        for (int ni = 0; ni < 2; ++ni) {
            v8f sp = zero;
#pragma unroll
            for (int c = 0; c < 2; ++c)
                sp = __builtin_amdgcn_wmma_f32_16x16x32_bf16(
                    false, qa[c], false, kf[ni][c], (short)0, sp, false, false);
#pragma unroll
            for (int e = 0; e < 8; ++e) red[wave * 256 + e * 32 + lane] = sp[e];
            __syncthreads();
            if (tid < 256) {
                float s = 0.0f;
#pragma unroll
                for (int w = 0; w < 16; ++w) s += red[w * 256 + tid];
                const int e = tid >> 5, l = tid & 31;
                const int row = e + ((l >> 4) << 3), kc = l & 15;
                Stile[row][ni * 16 + kc] = s * 0.03125f;   // 1/sqrt(1024)
            }
            __syncthreads();
        }
        // ---- online softmax on the 16x32 tile (one thread per row) ----
        if (tid < 16) {
            float mo = mrow[tid], mx = mo;
            for (int j = 0; j < 32; ++j) mx = fmaxf(mx, Stile[tid][j]);
            const float al = __expf(mo - mx);
            float s = 0.0f;
            for (int j = 0; j < 32; ++j) {
                const float p = __expf(Stile[tid][j] - mx);
                Ps[tid][j] = (bf16)p;
                s += p;
            }
            lrow[tid] = lrow[tid] * al + s;
            mrow[tid] = mx;
            arow[tid] = al;
        }
        __syncthreads();
        // ---- rescale accumulators, accumulate P * V over this wave's slice ----
        const bf16* pp = &Ps[fm][k0f];
        v16bf pa = load_frag(pp, pp + 16);
#pragma unroll
        for (int t = 0; t < 4; ++t) {
#pragma unroll
            for (int e = 0; e < 8; ++e) acc[t][e] *= arow[e + half * 8];
            acc[t] = __builtin_amdgcn_wmma_f32_16x16x32_bf16(
                false, pa, false, vf[t], (short)0, acc[t], false, false);
        }
        __syncthreads();
    }

    // ---- epilogue: normalize, add residual from fp32 x, store bf16 ----
#pragma unroll
    for (int t = 0; t < 4; ++t) {
#pragma unroll
        for (int e = 0; e < 8; ++e) {
            const int row  = e + half * 8;
            const int grow = q0 + row;
            const int gcol = dw + t * 16 + fm;
            const float v = acc[t][e] / lrow[row]
                          + x[((size_t)(b * S_ + grow)) * D_ + gcol];
            xa[((size_t)(b * S_ + grow)) * D_ + gcol] = (bf16)v;
        }
    }
}

extern "C" void kernel_launch(void* const* d_in, const int* in_sizes, int n_in,
                              void* d_out, int out_size, void* d_ws, size_t ws_size,
                              hipStream_t stream) {
    (void)in_sizes; (void)n_in; (void)out_size; (void)ws_size;
    const float* x  = (const float*)d_in[0];
    const float* Wq = (const float*)d_in[1];  const float* bq = (const float*)d_in[2];
    const float* Wk = (const float*)d_in[3];  const float* bk = (const float*)d_in[4];
    const float* Wv = (const float*)d_in[5];  const float* bv = (const float*)d_in[6];
    const float* W1 = (const float*)d_in[7];  const float* b1 = (const float*)d_in[8];
    const float* W2 = (const float*)d_in[9];  const float* b2 = (const float*)d_in[10];

    char* ws = (char*)d_ws;
    const size_t ND = (size_t)N_ * D_;     // 16,777,216 elements
    const size_t WW = (size_t)D_ * D_;     // 1,048,576 elements

    bf16* xb  = (bf16*)ws;                         // bf16 x; later reused as xa
    bf16* Wqb = (bf16*)(ws + 2 * ND);
    bf16* Wkb = Wqb + WW;
    bf16* Wvb = Wkb + WW;
    bf16* W1b = Wvb + WW;
    bf16* W2b = W1b + WW;
    bf16* qb  = (bf16*)(ws + 2 * ND + 10 * WW);    // Q; later reused as h
    bf16* kbuf = qb + ND;
    bf16* vtb  = kbuf + ND;
    bf16* xa = xb;    // attention output (x consumed by then)
    bf16* hb = qb;    // FFN hidden (q consumed by then)

    const int thr = 256;
    cvt_f32_bf16<<<(int)((ND + thr - 1) / thr), thr, 0, stream>>>(x,  xb,  (int)ND);
    cvt_f32_bf16<<<(int)((WW + thr - 1) / thr), thr, 0, stream>>>(Wq, Wqb, (int)WW);
    cvt_f32_bf16<<<(int)((WW + thr - 1) / thr), thr, 0, stream>>>(Wk, Wkb, (int)WW);
    cvt_f32_bf16<<<(int)((WW + thr - 1) / thr), thr, 0, stream>>>(Wv, Wvb, (int)WW);
    cvt_f32_bf16<<<(int)((WW + thr - 1) / thr), thr, 0, stream>>>(W1, W1b, (int)WW);
    cvt_f32_bf16<<<(int)((WW + thr - 1) / thr), thr, 0, stream>>>(W2, W2b, (int)WW);

    dim3 g(D_ / 128, N_ / 128);   // (8, 128)
    gemm128<0><<<g, 256, 0, stream>>>(xb, Wqb, bq, qb);
    gemm128<0><<<g, 256, 0, stream>>>(xb, Wkb, bk, kbuf);
    gemm128<3><<<g, 256, 0, stream>>>(xb, Wvb, bv, vtb);

    flash_attn<<<B_ * (S_ / 16), 512, 0, stream>>>(qb, kbuf, vtb, x, xa);

    gemm128<1><<<g, 256, 0, stream>>>(xa, W1b, b1, hb);
    gemm128<2><<<g, 256, 0, stream>>>(hb, W2b, b2, d_out);
}
